// CrossTrajectoryAttention_57939108823081
// MI455X (gfx1250) — matrix operand order repaired and checked by reading, not compiled, not run on hardware
//
#include <hip/hip_runtime.h>
#include <hip/hip_bf16.h>

// ---------------------------------------------------------------------------
// CrossTrajectoryAttention for MI455X (gfx1250, wave32, WMMA)
// B=2, S=1568, DIM=768, H=12, d=64, F=8, n=196 tokens/frame
// All GEMMs + stage-1 attention matmuls run on v_wmma_f32_16x16x32_bf16; stage 2 is scalar.
// GEMM uses async global->LDS double-buffered staging when the gfx1250
// async-LDS builtins are available (falls back to direct global->VGPR WMMA).
// ---------------------------------------------------------------------------

typedef __attribute__((ext_vector_type(16))) __bf16 v16bf;
typedef __attribute__((ext_vector_type(8)))  __bf16 v8bf;
typedef __attribute__((ext_vector_type(8)))  float  v8f;
typedef __attribute__((ext_vector_type(4)))  int    v4i;

#define BATCH 2
#define SEQ   1568
#define DIM   768
#define HEADS 12
#define HD    64
#define FRAMES 8
#define NTOK  196      // tokens per frame
#define NPAD  224      // padded per-frame token count (multiple of 32)
#define SCALE 0.125f   // 64^-0.5

#define AS1 __attribute__((address_space(1)))
#define AS3 __attribute__((address_space(3)))

#if defined(__AMDGCN__) && __has_builtin(__builtin_amdgcn_global_load_async_to_lds_b128)
#define USE_ASYNC_LDS 1
#else
#define USE_ASYNC_LDS 0
#endif

#if USE_ASYNC_LDS
#if __has_builtin(__builtin_amdgcn_s_wait_asynccnt)
#define WAIT_ASYNC0() __builtin_amdgcn_s_wait_asynccnt(0)
#else
#define WAIT_ASYNC0() asm volatile("s_wait_asynccnt 0" ::: "memory")
#endif
#endif

__device__ __forceinline__ v8f wmma_bf16(v16bf a, v16bf b, v8f c) {
  // D = A(16x32) x B(32x16) + C, f32 accum
  return __builtin_amdgcn_wmma_f32_16x16x32_bf16(
      /*neg_a=*/false, a, /*neg_b=*/false, b,
      /*c_mod=*/(short)0, c, /*reuse_a=*/false, /*reuse_b=*/false);
}

// A fragment: 16(M) x 32(K) bf16, row-major source with leading dim `ld`.
// Contiguous k order used here: lane<16 -> row=lane, K 0..15; lane>=16 -> row=lane-16, K 16..31
__device__ __forceinline__ v16bf load_a_frag(const __bf16* base, int ld) {
  int lane = threadIdx.x & 31;
  int row  = lane & 15;
  int kb   = (lane < 16) ? 0 : 16;
  const __bf16* p = base + (size_t)row * ld + kb;
  return *(const v16bf*)(p);
}

// B fragment: 32(K) x 16(N) bf16 from a transposed (N x K row-major) source.
// Contiguous k order used here: lane<16 -> col=lane, K 0..15; lane>=16 -> col=lane-16, K 16..31
__device__ __forceinline__ v16bf load_b_frag(const __bf16* base, int ld, int row) {
  int lane = threadIdx.x & 31;
  int kh   = (lane < 16) ? 0 : 16;
  return *(const v16bf*)(base + (size_t)row * ld + kh);
}

// ---------------------------------------------------------------------------
// Elementwise conversions
// ---------------------------------------------------------------------------
__global__ void f32_to_bf16_kernel(const float* __restrict__ in,
                                   __bf16* __restrict__ out, size_t n) {
  size_t i = (size_t)blockIdx.x * blockDim.x + threadIdx.x;
  if (i < n) out[i] = (__bf16)in[i];
}

// W (K x N, f32, row-major) -> Wt (N x K, bf16) so WMMA B-fragments are contiguous
__global__ void weight_transpose_bf16_kernel(const float* __restrict__ W,
                                             __bf16* __restrict__ Wt,
                                             int K, int N) {
  size_t i = (size_t)blockIdx.x * blockDim.x + threadIdx.x;
  size_t total = (size_t)K * N;
  if (i >= total) return;
  int n  = (int)(i % N);
  int kk = (int)(i / N);
  Wt[(size_t)n * K + kk] = (__bf16)W[i];
}

// ---------------------------------------------------------------------------
// Generic WMMA GEMM: C(MxN) = A(MxK) * Bt(NxK)^T,  bf16 in, f32 accum.
// Block = 128 threads = 4 waves, each wave owns a 32x32 tile -> 64x64 block.
// Async path: double-buffered 64x32 A/B tiles staged via
// global_load_async_to_lds_b128, consumed from LDS by WMMA fragments.
// MODE 0: f32 out (+ optional bias)
// MODE 1: bf16 out row-major
// MODE 2: kv split -> cols<768: K bf16 row-major (C0);
//         cols>=768: V transposed vT[b][h][d][f][NPAD] (C1)
// ---------------------------------------------------------------------------

#if USE_ASYNC_LDS
// Stage one 64(row) x 32(K) bf16 tile (4 KB) into LDS: 256 x 16B async loads,
// 128 threads -> 2 async b128 per thread per tile.
__device__ __forceinline__ void async_copy_tile(const __bf16* __restrict__ g,
                                                __bf16* l, int ldK, int k0) {
  int t = threadIdx.x;
  for (int u = t; u < 256; u += 128) {
    int row = u >> 2, seg = u & 3;
    const __bf16* gp = g + (size_t)row * ldK + k0 + seg * 8;
    __bf16* lp = l + row * 32 + seg * 8;
    __builtin_amdgcn_global_load_async_to_lds_b128(
        (AS1 v4i*)(AS1 void*)gp, (AS3 v4i*)(AS3 void*)lp, 0, 0);
  }
}
#endif

template <int MODE>
__global__ void gemm_wmma_kernel(const __bf16* __restrict__ A,
                                 const __bf16* __restrict__ Bt,
                                 void* __restrict__ C0, void* __restrict__ C1,
                                 const float* __restrict__ bias,
                                 int M, int N, int K) {
  int w  = threadIdx.x >> 5;
  int wm = w >> 1, wn = w & 1;
  int m0 = blockIdx.y * 64 + wm * 32;
  int n0 = blockIdx.x * 64 + wn * 32;

  v8f acc[2][2] = {};
  int rlo = (threadIdx.x & 31) & 15;

#if USE_ASYNC_LDS
  __shared__ __attribute__((aligned(32))) __bf16 Atile[2][64 * 32];
  __shared__ __attribute__((aligned(32))) __bf16 Btile[2][64 * 32];

  const __bf16* Ablk = A  + (size_t)(blockIdx.y * 64) * K;
  const __bf16* Bblk = Bt + (size_t)(blockIdx.x * 64) * K;

  async_copy_tile(Ablk, Atile[0], K, 0);
  async_copy_tile(Bblk, Btile[0], K, 0);

  int nsteps = K / 32;
  for (int idx = 0; idx < nsteps; ++idx) {
    int cur = idx & 1;
    WAIT_ASYNC0();        // this wave's async writes for buffer `cur` done
    __syncthreads();      // all waves' writes visible; other buffer free
    if (idx + 1 < nsteps) {
      async_copy_tile(Ablk, Atile[cur ^ 1], K, (idx + 1) * 32);
      async_copy_tile(Bblk, Btile[cur ^ 1], K, (idx + 1) * 32);
    }
    const __bf16* As = &Atile[cur][(wm * 32) * 32];
    const __bf16* Bs = &Btile[cur][(wn * 32) * 32];
    v16bf a0 = load_a_frag(As, 32);
    v16bf a1 = load_a_frag(As + 16 * 32, 32);
    v16bf b0 = load_b_frag(Bs, 32, rlo);
    v16bf b1 = load_b_frag(Bs + 16 * 32, 32, rlo);
    acc[0][0] = wmma_bf16(a0, b0, acc[0][0]);
    acc[0][1] = wmma_bf16(a0, b1, acc[0][1]);
    acc[1][0] = wmma_bf16(a1, b0, acc[1][0]);
    acc[1][1] = wmma_bf16(a1, b1, acc[1][1]);
  }
#else
  const __bf16* Abase = A  + (size_t)m0 * K;
  const __bf16* Bbase = Bt + (size_t)n0 * K;
  for (int k0 = 0; k0 < K; k0 += 32) {
    if (k0 + 256 < K) {  // gfx1250 global_prefetch hint
      __builtin_prefetch(Abase + k0 + 256, 0, 0);
      __builtin_prefetch(Bbase + k0 + 256, 0, 0);
    }
    v16bf a0 = load_a_frag(Abase + k0, K);
    v16bf a1 = load_a_frag(Abase + (size_t)16 * K + k0, K);
    v16bf b0 = load_b_frag(Bbase + k0, K, rlo);
    v16bf b1 = load_b_frag(Bbase + (size_t)16 * K + k0, K, rlo);
    acc[0][0] = wmma_bf16(a0, b0, acc[0][0]);
    acc[0][1] = wmma_bf16(a0, b1, acc[0][1]);
    acc[1][0] = wmma_bf16(a1, b0, acc[1][0]);
    acc[1][1] = wmma_bf16(a1, b1, acc[1][1]);
  }
#endif

  int lane  = threadIdx.x & 31;
  int cn    = lane & 15;
  int rbase = (lane < 16) ? 0 : 8;
  for (int mt = 0; mt < 2; ++mt)
    for (int nt = 0; nt < 2; ++nt) {
      v8f v = acc[mt][nt];
      for (int r = 0; r < 8; ++r) {
        int row = m0 + mt * 16 + rbase + r;
        int col = n0 + nt * 16 + cn;
        float val = v[r];
        if constexpr (MODE == 0) {
          ((float*)C0)[(size_t)row * N + col] = val + (bias ? bias[col] : 0.0f);
        } else if constexpr (MODE == 1) {
          ((__bf16*)C0)[(size_t)row * N + col] = (__bf16)val;
        } else {
          if (col < DIM) {
            ((__bf16*)C0)[(size_t)row * DIM + col] = (__bf16)val;
          } else {
            int c2 = col - DIM;
            int h = c2 >> 6, d = c2 & 63;
            int b = row / SEQ, m = row - b * SEQ;
            int f = m / NTOK, j = m - f * NTOK;
            ((__bf16*)C1)[((((size_t)b * HEADS + h) * HD + d) * FRAMES + f) * NPAD + j] =
                (__bf16)val;
          }
        }
      }
    }
}

// ---------------------------------------------------------------------------
// Stage-1 attention: one wave per (b, h, frame, 16-query tile).
// Scores(16x196) via WMMA -> softmax(196) in LDS -> P(16xNPAD, bf16) -> PV via WMMA.
// x out: [b][s][f][h*64+d] bf16.
// ---------------------------------------------------------------------------
__global__ void attn1_kernel(const __bf16* __restrict__ qb,
                             const __bf16* __restrict__ kb,
                             const __bf16* __restrict__ vT,
                             __bf16* __restrict__ x) {
  __shared__ float sc[16 * 208];
  __shared__ __attribute__((aligned(32))) __bf16 P[16 * NPAD];

  int lane = threadIdx.x;
  int s0 = blockIdx.x * 16;
  int f  = blockIdx.y;
  int bh = blockIdx.z;
  int b = bh / HEADS, h = bh % HEADS;

  // Q tile: 16 queries x 64 (head dim)
  const __bf16* qbase = qb + ((size_t)(b * SEQ + s0)) * DIM + h * HD;
  v16bf a0 = load_a_frag(qbase, DIM);       // d 0..31
  v16bf a1 = load_a_frag(qbase + 32, DIM);  // d 32..63

  // Scores: 13 token tiles of 16 (196 -> 208, last tile row-clamped)
  const __bf16* kbase = kb + ((size_t)(b * SEQ + f * NTOK)) * DIM + h * HD;
  int rb = (lane < 16) ? 0 : 8;
  for (int t = 0; t < 13; ++t) {
    int tok = t * 16 + (lane & 15);
    if (tok > NTOK - 1) tok = NTOK - 1;     // clamp: garbage cols masked later
    v16bf b0 = load_b_frag(kbase, DIM, tok);
    v16bf b1 = load_b_frag(kbase + 32, DIM, tok);
    v8f s = {};
    s = wmma_bf16(a0, b0, s);
    s = wmma_bf16(a1, b1, s);
    for (int r = 0; r < 8; ++r)
      sc[(rb + r) * 208 + t * 16 + (lane & 15)] = s[r];
  }
  __syncthreads();

  // Softmax over 196 per row; masked columns -> exact 0 so padded V is annihilated
  if (lane < 16) {
    float* row = sc + lane * 208;
    float mx = -1e30f;
    for (int j = 0; j < NTOK; ++j) {
      float v = row[j] * SCALE;
      row[j] = v;
      mx = fmaxf(mx, v);
    }
    float sum = 0.0f;
    for (int j = 0; j < NTOK; ++j) {
      float e = __expf(row[j] - mx);
      row[j] = e;
      sum += e;
    }
    float inv = 1.0f / sum;
    __bf16* pr = P + lane * NPAD;
    for (int j = 0; j < NTOK; ++j) pr[j] = (__bf16)(row[j] * inv);
    for (int j = NTOK; j < NPAD; ++j) pr[j] = (__bf16)0.0f;
  }
  __syncthreads();

  // PV: out(16x64) = P(16xNPAD) x V(NPADx64); V stored transposed+padded
  v8f o[4] = {};
  const __bf16* vbase = vT + (((size_t)(b * HEADS + h)) * HD * FRAMES + f) * NPAD;
  for (int kk = 0; kk < NPAD / 32; ++kk) {
    v16bf pa = load_a_frag(P + kk * 32, NPAD);
    for (int nt = 0; nt < 4; ++nt) {
      v16bf vb = load_b_frag(vbase + kk * 32, FRAMES * NPAD, nt * 16 + (lane & 15));
      o[nt] = wmma_bf16(pa, vb, o[nt]);
    }
  }
  int cn = lane & 15;
  for (int nt = 0; nt < 4; ++nt)
    for (int r = 0; r < 8; ++r) {
      int srow = s0 + rb + r;
      int col  = h * HD + nt * 16 + cn;
      x[(((size_t)(b * SEQ + srow)) * FRAMES + f) * DIM + col] = (__bf16)o[nt][r];
    }
}

// x_diag[b, s, :] = x[b, s, f=s/196, :]
__global__ void diag_gather_kernel(const __bf16* __restrict__ x,
                                   __bf16* __restrict__ xd) {
  size_t i = (size_t)blockIdx.x * blockDim.x + threadIdx.x;
  const size_t total = (size_t)BATCH * SEQ * DIM;
  if (i >= total) return;
  int c = (int)(i % DIM);
  size_t bs = i / DIM;
  int s = (int)(bs % SEQ);
  int f = s / NTOK;
  xd[i] = x[(bs * FRAMES + f) * DIM + c];
}

// ---------------------------------------------------------------------------
// Stage-2 attention (f=8, tiny): one wave per (b, s, h).
// ---------------------------------------------------------------------------
__global__ void attn2_kernel(const float* __restrict__ q2,
                             const __bf16* __restrict__ kv2,
                             __bf16* __restrict__ outpre,
                             float* __restrict__ attn2out) {
  int s  = blockIdx.x;
  int bh = blockIdx.y;
  int b = bh / HEADS, h = bh % HEADS;
  int lane = threadIdx.x;

  const float*  qp    = q2  + ((size_t)(b * SEQ + s)) * DIM + h * HD;
  const __bf16* kvrow = kv2 + ((size_t)(b * SEQ + s)) * FRAMES * (2 * DIM);

  float dots[FRAMES];
  for (int f = 0; f < FRAMES; ++f) {
    const __bf16* kp = kvrow + (size_t)f * (2 * DIM) + h * HD;
    float acc = 0.0f;
    for (int d = 0; d < HD; ++d) acc += qp[d] * (float)kp[d];
    dots[f] = acc * SCALE;
  }
  float mx = dots[0];
  for (int f = 1; f < FRAMES; ++f) mx = fmaxf(mx, dots[f]);
  float sum = 0.0f;
  for (int f = 0; f < FRAMES; ++f) { dots[f] = __expf(dots[f] - mx); sum += dots[f]; }
  float inv = 1.0f / sum;
  for (int f = 0; f < FRAMES; ++f) dots[f] *= inv;

  if (lane < FRAMES)
    attn2out[(((size_t)(b * HEADS + h)) * SEQ + s) * FRAMES + lane] = dots[lane];

  for (int d = lane; d < HD; d += 32) {
    float o = 0.0f;
    for (int f = 0; f < FRAMES; ++f)
      o += dots[f] * (float)kvrow[(size_t)f * (2 * DIM) + DIM + h * HD + d];
    outpre[((size_t)(b * SEQ + s)) * DIM + h * HD + d] = (__bf16)o;
  }
}

// ---------------------------------------------------------------------------
// Host launcher
// ---------------------------------------------------------------------------
extern "C" void kernel_launch(void* const* d_in, const int* in_sizes, int n_in,
                              void* d_out, int out_size, void* d_ws, size_t ws_size,
                              hipStream_t stream) {
  const float* xq    = (const float*)d_in[0];
  const float* xk    = (const float*)d_in[1];
  const float* Wq    = (const float*)d_in[2];
  const float* Wkv   = (const float*)d_in[3];
  const float* Wpq   = (const float*)d_in[4];
  const float* Wpkv  = (const float*)d_in[5];
  const float* Wproj = (const float*)d_in[6];
  const float* bproj = (const float*)d_in[7];

  const size_t ACT = (size_t)BATCH * SEQ * DIM;          // 2,408,448 elems
  const size_t XEL = ACT * FRAMES;                       // x: b,s,f,dim
  const size_t KV2 = (size_t)BATCH * SEQ * FRAMES * 2 * DIM;
  const size_t VTE = (size_t)BATCH * HEADS * HD * FRAMES * NPAD;

  unsigned char* ws = (unsigned char*)d_ws;
  size_t off = 0;
  auto take = [&](size_t bytes) -> unsigned char* {
    unsigned char* p = ws + off;
    off = (off + bytes + 255) & ~(size_t)255;
    return p;
  };

  __bf16* xqb    = (__bf16*)take(ACT * 2);
  __bf16* xkb    = (__bf16*)take(ACT * 2);
  __bf16* WqT    = (__bf16*)take((size_t)DIM * DIM * 2);
  __bf16* WkvT   = (__bf16*)take((size_t)DIM * 2 * DIM * 2);
  __bf16* WpqT   = (__bf16*)take((size_t)DIM * DIM * 2);
  __bf16* WpkvT  = (__bf16*)take((size_t)DIM * 2 * DIM * 2);
  __bf16* WprojT = (__bf16*)take((size_t)DIM * DIM * 2);
  __bf16* qb     = (__bf16*)take(ACT * 2);
  __bf16* kb     = (__bf16*)take(ACT * 2);
  __bf16* vT     = (__bf16*)take(VTE * 2);
  __bf16* xb     = (__bf16*)take(XEL * 2);
  __bf16* xdiag  = (__bf16*)take(ACT * 2);
  float*  q2     = (float*)take(ACT * 4);
  __bf16* kv2    = (__bf16*)take(KV2 * 2);
  __bf16* outpre = (__bf16*)take(ACT * 2);

  float* out_main  = (float*)d_out;
  float* out_attn2 = (float*)d_out + ACT;

  const int M  = BATCH * SEQ;          // 3136 (= 49 * 64)
  const int M2 = BATCH * SEQ * FRAMES; // 25088 (= 392 * 64)

  // 1) fp32 -> bf16 activations; weights -> transposed bf16
  f32_to_bf16_kernel<<<(unsigned)((ACT + 255) / 256), 256, 0, stream>>>(xq, xqb, ACT);
  f32_to_bf16_kernel<<<(unsigned)((ACT + 255) / 256), 256, 0, stream>>>(xk, xkb, ACT);
  {
    size_t w1 = (size_t)DIM * DIM, w2 = (size_t)DIM * 2 * DIM;
    weight_transpose_bf16_kernel<<<(unsigned)((w1 + 255) / 256), 256, 0, stream>>>(Wq,    WqT,    DIM, DIM);
    weight_transpose_bf16_kernel<<<(unsigned)((w2 + 255) / 256), 256, 0, stream>>>(Wkv,   WkvT,   DIM, 2 * DIM);
    weight_transpose_bf16_kernel<<<(unsigned)((w1 + 255) / 256), 256, 0, stream>>>(Wpq,   WpqT,   DIM, DIM);
    weight_transpose_bf16_kernel<<<(unsigned)((w2 + 255) / 256), 256, 0, stream>>>(Wpkv,  WpkvT,  DIM, 2 * DIM);
    weight_transpose_bf16_kernel<<<(unsigned)((w1 + 255) / 256), 256, 0, stream>>>(Wproj, WprojT, DIM, DIM);
  }

  // 2) projections: q = xq@Wq (bf16 out); kv = xk@Wkv (K row-major, V transposed)
  gemm_wmma_kernel<1><<<dim3(DIM / 64, M / 64), 128, 0, stream>>>(
      xqb, WqT, qb, nullptr, nullptr, M, DIM, DIM);
  gemm_wmma_kernel<2><<<dim3(2 * DIM / 64, M / 64), 128, 0, stream>>>(
      xkb, WkvT, kb, vT, nullptr, M, 2 * DIM, DIM);

  // 3) stage-1 attention -> x (b,s,f,dim) bf16
  attn1_kernel<<<dim3(SEQ / 16, FRAMES, BATCH * HEADS), 32, 0, stream>>>(qb, kb, vT, xb);

  // 4) diagonal gather + stage-2 projections
  diag_gather_kernel<<<(unsigned)((ACT + 255) / 256), 256, 0, stream>>>(xb, xdiag);
  gemm_wmma_kernel<0><<<dim3(DIM / 64, M / 64), 128, 0, stream>>>(
      xdiag, WpqT, q2, nullptr, nullptr, M, DIM, DIM);
  gemm_wmma_kernel<1><<<dim3(2 * DIM / 64, M2 / 64), 128, 0, stream>>>(
      xb, WpkvT, kv2, nullptr, nullptr, M2, 2 * DIM, DIM);

  // 5) stage-2 attention (writes attn2 output) + final projection with bias
  attn2_kernel<<<dim3(SEQ, BATCH * HEADS), 32, 0, stream>>>(q2, kv2, outpre, out_attn2);
  gemm_wmma_kernel<0><<<dim3(DIM / 64, M / 64), 128, 0, stream>>>(
      outpre, WprojT, out_main, nullptr, bproj, M, DIM, DIM);
}
